// HumanVAttention_60352880443493
// MI455X (gfx1250) — matrix operand order repaired and checked by reading, not compiled, not run on hardware
//
#include <hip/hip_runtime.h>
#include <hip/hip_bf16.h>

// ---------------------------------------------------------------------------
// Problem constants (compile-time, from the reference)
// ---------------------------------------------------------------------------
#define S_LEN 4096
#define HID   2048
#define NH    16
#define NKV   4
#define HD    128
#define BLK   64
#define NB    (S_LEN / BLK)     // 64
#define ROT   64
#define QDIM  (NH * HD)         // 2048
#define KVDIM (NKV * HD)        // 512
#define SCALE 0.08838834764831845f   // 128^-0.5
#define NEGF  -1000000000.0f

typedef __attribute__((ext_vector_type(16))) __bf16 v16bf;
typedef __attribute__((ext_vector_type(8)))  float  v8f;

// ---------------------------------------------------------------------------
// CDNA5 async global->LDS copy (16B per lane), tracked by ASYNCcnt.
// LDS operand: low 32 bits of a generic pointer to __shared__ is the LDS
// byte offset (ISA 10.2 aperture layout).
// ---------------------------------------------------------------------------
__device__ __forceinline__ void async_g2l_b128(const void* gaddr, void* lds) {
  unsigned l = (unsigned)(size_t)lds;
  asm volatile("global_load_async_to_lds_b128 %0, %1, off"
               :: "v"(l), "v"(gaddr) : "memory");
}
__device__ __forceinline__ void wait_asynccnt0() {
  asm volatile("s_wait_asynccnt 0x0" ::: "memory");
}

// ---------------------------------------------------------------------------
// WMMA fragment loaders (CDNA5 ISA 7.12.2 layouts, wave32)
//   A (16x32 bf16): lane -> M (lane&15); VGPR j holds K pair:
//     K = (j<4 ? 2j : 16+2(j-4)) + 8*(lane>>4) + (e&1)
//   B (32x16 bf16): lane -> N (lane&15); element e -> K, same K order as A
// ---------------------------------------------------------------------------
__device__ __forceinline__ v16bf load_a_frag(const __bf16* row, int half) {
  union { v16bf v; unsigned u[8]; } r;
#pragma unroll
  for (int j = 0; j < 8; ++j) {
    int kb = ((j < 4) ? (2 * j) : (16 + 2 * (j - 4))) + 8 * half;
    r.u[j] = *(const unsigned*)(row + kb);   // pair (kb, kb+1), 4B aligned
  }
  return r.v;
}

__device__ __forceinline__ v16bf load_b_frag(const __bf16* p, int ld, int half) {
  union { v16bf v; __bf16 e[16]; } r;
#pragma unroll
  for (int s = 0; s < 16; ++s) {
    int k = ((s < 8) ? s : (8 + s)) + 8 * half;
    r.e[s] = p[(size_t)k * ld];
  }
  return r.v;
}

__device__ __forceinline__ v8f wmma_bf16(v16bf a, v16bf b, v8f c) {
  return __builtin_amdgcn_wmma_f32_16x16x32_bf16(false, a, false, b,
                                                 (short)0, c, false, false);
}

// ---------------------------------------------------------------------------
// fp32 -> bf16 conversion, 4 elements / thread
// ---------------------------------------------------------------------------
__global__ void k_cvt_bf16(const float* __restrict__ s, __bf16* __restrict__ d,
                           int n4) {
  int i = blockIdx.x * blockDim.x + threadIdx.x;
  if (i >= n4) return;
  float4 v = ((const float4*)s)[i];
  union { __bf16 h[4]; unsigned long long u; } r;
  r.h[0] = (__bf16)v.x; r.h[1] = (__bf16)v.y;
  r.h[2] = (__bf16)v.z; r.h[3] = (__bf16)v.w;
  ((unsigned long long*)d)[i] = r.u;
}

// ---------------------------------------------------------------------------
// Generic WMMA GEMM: C[M,N] = A[M,K] @ B[K,N], A/B bf16 row-major.
// Block = 256 thr (8 waves). Tile 128(M) x 64(N), K-step 64.
// LDS staged via async global->LDS (ASYNCcnt), 8 WMMAs / wave / step.
// ---------------------------------------------------------------------------
template <typename OutT>
__global__ __launch_bounds__(256)
void k_gemm(const __bf16* __restrict__ A, const __bf16* __restrict__ B,
            OutT* __restrict__ C, int M, int N, int K) {
  __shared__ __align__(16) __bf16 As[128 * 64];   // 16 KB
  __shared__ __align__(16) __bf16 Bs[64 * 64];    //  8 KB
  const int tid  = threadIdx.x;
  const int lane = tid & 31, wave = tid >> 5;
  const int half = lane >> 4, mlane = lane & 15;
  const int row0 = blockIdx.y * 128;
  const int col0 = blockIdx.x * 64;

  v8f acc[4] = {};

  for (int k0 = 0; k0 < K; k0 += 64) {
    // stage A tile 128x64 bf16: 1024 x 16B chunks, 4 per thread
#pragma unroll
    for (int i = 0; i < 4; ++i) {
      int idx = tid + i * 256;
      int r = idx >> 3, ce = (idx & 7) * 8;       // 8 chunks (of 8 elems) / row
      async_g2l_b128(A + (size_t)(row0 + r) * K + k0 + ce, As + r * 64 + ce);
    }
    // stage B tile 64x64 bf16: 512 x 16B chunks, 2 per thread
#pragma unroll
    for (int i = 0; i < 2; ++i) {
      int idx = tid + i * 256;
      int r = idx >> 3, ce = (idx & 7) * 8;
      async_g2l_b128(B + (size_t)(k0 + r) * N + col0 + ce, Bs + r * 64 + ce);
    }
    wait_asynccnt0();      // drain this wave's async copies
    __syncthreads();       // make all waves' copies visible

#pragma unroll
    for (int ks = 0; ks < 2; ++ks) {
      v16bf af = load_a_frag(As + (wave * 16 + mlane) * 64 + ks * 32, half);
#pragma unroll
      for (int t = 0; t < 4; ++t) {
        v16bf bf_ = load_b_frag(Bs + (ks * 32) * 64 + t * 16 + mlane, 64, half);
        acc[t] = wmma_bf16(af, bf_, acc[t]);
      }
    }
    __syncthreads();
  }

#pragma unroll
  for (int t = 0; t < 4; ++t)
#pragma unroll
    for (int r = 0; r < 8; ++r) {
      int rr = row0 + wave * 16 + r + 8 * half;
      int cc = col0 + t * 16 + mlane;
      C[(size_t)rr * N + cc] = (OutT)acc[t][r];
    }
}

// ---------------------------------------------------------------------------
// RoPE on Q in place: layout [S][NH*HD]; rotate first 64 dims of each head.
// One thread per (s, h, p), p in 0..31 handles pair (p, p+32).
// ---------------------------------------------------------------------------
__global__ void k_rope_q(__bf16* __restrict__ q, const float* __restrict__ cs,
                         const float* __restrict__ sn) {
  int i = blockIdx.x * blockDim.x + threadIdx.x;   // S*NH*32 total
  int p = i & 31;
  int h = (i >> 5) & (NH - 1);
  int s = i >> 9;
  __bf16* base = q + (size_t)s * QDIM + h * HD;
  float x0 = (float)base[p], x1 = (float)base[p + 32];
  float c0 = cs[s * ROT + p], c1 = cs[s * ROT + p + 32];
  float s0 = sn[s * ROT + p], s1 = sn[s * ROT + p + 32];
  base[p]      = (__bf16)(x0 * c0 - x1 * s0);   // rot_half: -x[d+32] for d<32
  base[p + 32] = (__bf16)(x1 * c1 + x0 * s1);   //           +x[d-32] for d>=32
}

// ---------------------------------------------------------------------------
// RoPE on K fused with transpose: Kin [S][KVDIM] -> Kt [KVDIM][S]
// grid (S/256, KVDIM), block 256. Coalesced writes along s.
// ---------------------------------------------------------------------------
__global__ void k_rope_kt(const __bf16* __restrict__ kin,
                          __bf16* __restrict__ kt,
                          const float* __restrict__ cs,
                          const float* __restrict__ sn) {
  int s = blockIdx.x * blockDim.x + threadIdx.x;   // 0..S-1
  int rowc = blockIdx.y;                           // kvh*HD + d
  int d = rowc & (HD - 1);
  size_t base = (size_t)s * KVDIM + rowc;
  float x = (float)kin[base];
  float val;
  if (d < 32)
    val = x * cs[s * ROT + d] - (float)kin[base + 32] * sn[s * ROT + d];
  else if (d < ROT)
    val = x * cs[s * ROT + d] + (float)kin[base - 32] * sn[s * ROT + d];
  else
    val = x;
  kt[(size_t)rowc * S_LEN + s] = (__bf16)val;
}

// ---------------------------------------------------------------------------
// Block-sparse flash attention. Grid (NH, NB), block 128 (4 waves).
// Wave w handles queries [bi*64 + w*16, +16). Online softmax over selected
// key blocks; score & PV matmuls via WMMA bf16.
// ---------------------------------------------------------------------------
__global__ __launch_bounds__(128)
void k_attn(const __bf16* __restrict__ Qb, const __bf16* __restrict__ Kt,
            const __bf16* __restrict__ Vb, const float* __restrict__ amask,
            __bf16* __restrict__ ctx) {
  __shared__ __align__(16) __bf16 Qs[64][128];
  __shared__ __align__(16) __bf16 Ps[4][16][64];

  const int h   = blockIdx.x;
  const int bi  = blockIdx.y;
  const int kvh = h >> 2;
  const int lane = threadIdx.x & 31, wave = threadIdx.x >> 5;
  const int half = lane >> 4, nlane = lane & 15;

  // stage this wave's own 16 Q rows via async copies (wave-private region,
  // so s_wait_asynccnt alone suffices -- no workgroup barrier needed)
  {
    const __bf16* src = Qb + (size_t)(bi * BLK + wave * 16) * QDIM + h * HD;
#pragma unroll
    for (int i = 0; i < 8; ++i) {
      int idx = lane + i * 32;                 // 256 x 16B chunks
      int r = idx >> 4, ce = (idx & 15) * 8;   // 16 chunks / 128-elem row
      async_g2l_b128(src + (size_t)r * QDIM + ce, &Qs[wave * 16 + r][ce]);
    }
    wait_asynccnt0();
  }

  // static block selection (local window + strided globals + block 0)
  int sel[8]; int nsel = 0;
  {
    unsigned long long msk = 1ull;                       // always block 0
    for (int i = bi - 3; i <= bi; ++i) if (i >= 0) msk |= 1ull << i;
    int ngc = bi / 4 + 1;
    int st = (ngc > 2) ? (ngc - 2) : 0;
    for (int t = st; t < ngc; ++t) msk |= 1ull << (t * 4);
    for (int i = 0; i <= bi; ++i) if ((msk >> i) & 1) sel[nsel++] = i;
  }

  float rmax[8], rsum[8];
  v8f ovac[8] = {};
#pragma unroll
  for (int r = 0; r < 8; ++r) { rmax[r] = -3.0e38f; rsum[r] = 0.0f; }

  for (int jj = 0; jj < nsel; ++jj) {
    const int blk = sel[jj];

    // ---- scores: 16 queries x 64 keys, 4 N-tiles ----
    v8f sc[4];
#pragma unroll
    for (int nt = 0; nt < 4; ++nt) {
      v8f a = {};
#pragma unroll
      for (int ks = 0; ks < 4; ++ks) {
        v16bf af = load_a_frag(&Qs[wave * 16 + nlane][ks * 32], half);
        v16bf bf_ = load_b_frag(
            Kt + (size_t)(kvh * HD + ks * 32) * S_LEN + blk * BLK + nt * 16 + nlane, S_LEN, half);
        a = wmma_bf16(af, bf_, a);
      }
      int keyc = blk * BLK + nt * 16 + nlane;
      float addm = (1.0f - amask[keyc]) * NEGF;
#pragma unroll
      for (int r = 0; r < 8; ++r) {
        float v = a[r] * SCALE + addm;
        if (blk == bi) {
          int qrow = wave * 16 + r + 8 * half;
          int krow = nt * 16 + nlane;
          if (krow > qrow) v = NEGF;           // causal intra-block
        }
        a[r] = v;
      }
      sc[nt] = a;
    }

    // ---- online softmax update ----
#pragma unroll
    for (int r = 0; r < 8; ++r) {
      float v = fmaxf(fmaxf(sc[0][r], sc[1][r]), fmaxf(sc[2][r], sc[3][r]));
#pragma unroll
      for (int o = 1; o < 16; o <<= 1) v = fmaxf(v, __shfl_xor(v, o, 32));
      float nm = fmaxf(rmax[r], v);
      float scl = __expf(rmax[r] - nm);
      rmax[r] = nm;
      rsum[r] *= scl;
#pragma unroll
      for (int n = 0; n < 8; ++n) ovac[n][r] *= scl;
    }

    // ---- P = exp(sc - rmax) -> LDS (C layout -> A layout) ----
#pragma unroll
    for (int nt = 0; nt < 4; ++nt)
#pragma unroll
      for (int r = 0; r < 8; ++r) {
        float p = __expf(sc[nt][r] - rmax[r]);
        rsum[r] += p;                          // per-lane partial row sum
        Ps[wave][r + 8 * half][nt * 16 + nlane] = (__bf16)p;
      }
    asm volatile("" ::: "memory");             // same-wave DS ops are in-order

    // ---- O += P @ V ----
#pragma unroll
    for (int kc = 0; kc < 2; ++kc) {
      v16bf pf = load_a_frag(&Ps[wave][nlane][kc * 32], half);
      const __bf16* vrow =
          Vb + (size_t)(blk * BLK + kc * 32) * KVDIM + kvh * HD + nlane;
#pragma unroll
      for (int n = 0; n < 8; ++n) {
        v16bf vf = load_b_frag(vrow + n * 16, KVDIM, half);
        ovac[n] = wmma_bf16(pf, vf, ovac[n]);
      }
    }
  }

  // ---- finalize: full row sums, normalize, store ctx bf16 ----
#pragma unroll
  for (int r = 0; r < 8; ++r) {
    float v = rsum[r];
#pragma unroll
    for (int o = 1; o < 16; o <<= 1) v += __shfl_xor(v, o, 32);
    rsum[r] = 1.0f / v;
  }
#pragma unroll
  for (int n = 0; n < 8; ++n)
#pragma unroll
    for (int r = 0; r < 8; ++r) {
      int q = bi * BLK + wave * 16 + r + 8 * half;
      ctx[(size_t)q * QDIM + h * HD + n * 16 + nlane] =
          (__bf16)(ovac[n][r] * rsum[r]);
    }
}

// ---------------------------------------------------------------------------
// Host side
// ---------------------------------------------------------------------------
extern "C" void kernel_launch(void* const* d_in, const int* in_sizes, int n_in,
                              void* d_out, int out_size, void* d_ws,
                              size_t ws_size, hipStream_t stream) {
  const float* hidden = (const float*)d_in[0];
  const float* cosb   = (const float*)d_in[1];
  const float* sinb   = (const float*)d_in[2];
  const float* Wq     = (const float*)d_in[3];
  const float* Wk     = (const float*)d_in[4];
  const float* Wv     = (const float*)d_in[5];
  const float* Wo     = (const float*)d_in[6];
  const float* amask  = (const float*)d_in[7];
  float* out = (float*)d_out;

  char* ws = (char*)d_ws;
  size_t off = 0;
  auto alloc = [&](size_t elems) {
    void* p = ws + off;
    off = (off + elems * sizeof(__bf16) + 255) & ~(size_t)255;
    return (__bf16*)p;
  };
  __bf16* Xb  = alloc((size_t)S_LEN * HID);
  __bf16* Wqb = alloc((size_t)HID * QDIM);
  __bf16* Wkb = alloc((size_t)HID * KVDIM);
  __bf16* Wvb = alloc((size_t)HID * KVDIM);
  __bf16* Wob = alloc((size_t)QDIM * HID);
  __bf16* Qb  = alloc((size_t)S_LEN * QDIM);
  __bf16* Kb  = alloc((size_t)S_LEN * KVDIM);
  __bf16* Vb  = alloc((size_t)S_LEN * KVDIM);
  __bf16* Ktb = alloc((size_t)KVDIM * S_LEN);
  __bf16* Ctx = alloc((size_t)S_LEN * QDIM);

  auto cvt = [&](const float* s, __bf16* d, size_t n) {
    int n4 = (int)(n / 4);
    k_cvt_bf16<<<(n4 + 255) / 256, 256, 0, stream>>>(s, d, n4);
  };
  cvt(hidden, Xb, (size_t)S_LEN * HID);
  cvt(Wq, Wqb, (size_t)HID * QDIM);
  cvt(Wk, Wkb, (size_t)HID * KVDIM);
  cvt(Wv, Wvb, (size_t)HID * KVDIM);
  cvt(Wo, Wob, (size_t)QDIM * HID);

  // QKV projections
  k_gemm<__bf16><<<dim3(QDIM / 64, S_LEN / 128), 256, 0, stream>>>(
      Xb, Wqb, Qb, S_LEN, QDIM, HID);
  k_gemm<__bf16><<<dim3(KVDIM / 64, S_LEN / 128), 256, 0, stream>>>(
      Xb, Wkb, Kb, S_LEN, KVDIM, HID);
  k_gemm<__bf16><<<dim3(KVDIM / 64, S_LEN / 128), 256, 0, stream>>>(
      Xb, Wvb, Vb, S_LEN, KVDIM, HID);

  // RoPE
  k_rope_q<<<(S_LEN * NH * 32) / 256, 256, 0, stream>>>(Qb, cosb, sinb);
  k_rope_kt<<<dim3(S_LEN / 256, KVDIM), 256, 0, stream>>>(Kb, Ktb, cosb, sinb);

  // Block-sparse attention
  k_attn<<<dim3(NH, NB), 128, 0, stream>>>(Qb, Ktb, Vb, amask, Ctx);

  // Output projection -> f32
  k_gemm<float><<<dim3(HID / 64, S_LEN / 128), 256, 0, stream>>>(
      Ctx, Wob, out, S_LEN, HID, QDIM);
}
